// MultiHeadAttention_31447750541567
// MI455X (gfx1250) — matrix operand order repaired and checked by reading, not compiled, not run on hardware
//
#include <hip/hip_runtime.h>

// ---------- types ----------
typedef __bf16 bf16;
typedef __bf16 bf16x4  __attribute__((ext_vector_type(4)));
typedef __bf16 bf16x8  __attribute__((ext_vector_type(8)));
typedef __bf16 v16bf   __attribute__((ext_vector_type(16)));
typedef float  v8f     __attribute__((ext_vector_type(8)));
typedef unsigned int u32;
typedef u32 u32x4 __attribute__((ext_vector_type(4)));
typedef u32 u32x8 __attribute__((ext_vector_type(8)));

// problem sizes
#define BB   4
#define SSQ  4096
#define DDm  1024
#define HHd  16
#define DKK  64

// LDS slab geometry for TDM-staged B tiles: 128 N-rows x 64 K-elems bf16,
// DMA-padded 4 DWORDs per 32 DWORDs -> row pitch 72 elems (144B, conflict-free)
#define BROWS   128
#define BKS     64
#define BPITCH  72
#define SLAB_BYTES (BROWS * BPITCH * 2)

__device__ inline v16bf cat8(bf16x8 lo, bf16x8 hi) {
    return __builtin_shufflevector(lo, hi, 0,1,2,3,4,5,6,7,8,9,10,11,12,13,14,15);
}

__device__ inline v16bf f32x16_to_bf16(float4 f0, float4 f1, float4 f2, float4 f3) {
    v16bf t;
    t[0]=(bf16)f0.x;  t[1]=(bf16)f0.y;  t[2]=(bf16)f0.z;  t[3]=(bf16)f0.w;
    t[4]=(bf16)f1.x;  t[5]=(bf16)f1.y;  t[6]=(bf16)f1.z;  t[7]=(bf16)f1.w;
    t[8]=(bf16)f2.x;  t[9]=(bf16)f2.y;  t[10]=(bf16)f2.z; t[11]=(bf16)f2.w;
    t[12]=(bf16)f3.x; t[13]=(bf16)f3.y; t[14]=(bf16)f3.z; t[15]=(bf16)f3.w;
    return t;
}

__device__ inline void wait_tensor0() {
#if __has_builtin(__builtin_amdgcn_s_wait_tensorcnt)
    __builtin_amdgcn_s_wait_tensorcnt(0);
#else
    asm volatile("s_wait_tensorcnt 0x0" ::: "memory");
#endif
}

// TDM: DMA a 128x64 bf16 slab of W (row-major [N,K], K=1024) into LDS with padding.
// D# per CDNA5 ISA 8.3/8.4: group0 = {flags/count, lds_addr, global_addr, type},
// group1 = {mask/data_size/pad, tensor dims, tile dims, strides}.
__device__ inline void tdm_load_b_slab(const bf16* gsrc, u32 ldsOff) {
    const unsigned long long ga = (unsigned long long)(size_t)gsrc;
    u32x4 g0;
    g0[0] = 1u;                                        // count=1, user mode
    g0[1] = ldsOff;                                    // lds_addr (bytes)
    g0[2] = (u32)ga;                                   // global_addr[31:0]
    g0[3] = ((u32)(ga >> 32) & 0x01FFFFFFu) | (2u << 30);  // addr[56:32] | type=2
    u32x8 g1;
    g1[0] = (1u << 16)        // data_size = 2 bytes
          | (1u << 20)        // pad_enable
          | (4u << 22)        // pad_interval: every 32 DWORDs
          | (3u << 25);       // pad_amount: 4 DWORDs
    g1[1] = (1024u & 0xFFFFu) << 16;                  // tensor_dim0[15:0] (K)
    g1[2] = (1024u >> 16) | ((1024u & 0xFFFFu) << 16); // tensor_dim0[31:16] | tensor_dim1[15:0] (N)
    g1[3] = (1024u >> 16) | ((u32)BKS << 16);          // tensor_dim1[31:16] | tile_dim0 = 64
    g1[4] = (u32)BROWS;                                // tile_dim1 = 128 (tile_dim2 = 0)
    g1[5] = 1024u;                                     // tensor_dim0_stride[31:0] = K
    g1[6] = 0u;                                        // stride[47:32]=0, dim1_stride lo = 0
    g1[7] = 0u;
    asm volatile("tensor_load_to_lds %0, %1" :: "s"(g0), "s"(g1) : "memory");
}

// ---------- weight f32 -> bf16 ----------
__global__ __launch_bounds__(256)
void cvt_f32_bf16(const float* __restrict__ src, bf16* __restrict__ dst, int n) {
    int idx = (blockIdx.x * blockDim.x + threadIdx.x) * 4;
    if (idx + 3 < n) {
        float4 f = *(const float4*)(src + idx);
        bf16x4 o;
        o[0]=(bf16)f.x; o[1]=(bf16)f.y; o[2]=(bf16)f.z; o[3]=(bf16)f.w;
        *(bf16x4*)(dst + idx) = o;
    }
}

// ---------- GEMM: Out[M,N] = A[M,K] @ W[N,K]^T + bias ----------
// Block = 256 threads = 8 waves; block tile 128x128; wave tile 32(M) x 64(N).
// W slabs staged to LDS by the Tensor Data Mover, double-buffered on TENSORcnt.
// Fragment layouts (CDNA5 ISA 7.12.2, bf16 16x16x32):
//   A: lane(l<16): M=l, V0-3 = K(half*8+0..7), V4-7 = K(16+half*8+0..7)
//   B: lane: N=l&15, V0-3 = K(half*8+0..7), V4-7 = K(16+half*8+0..7)
//   C: vgpr j, lane l: M = j + 8*(l>>4), N = l&15
template<bool A_F32, bool OUT_F32>
__global__ __launch_bounds__(256)
void gemm_wmma_bias(const void* __restrict__ Aptr,
                    const bf16* __restrict__ W,
                    const float* __restrict__ bias,
                    void* __restrict__ Out,
                    int M, int N, int K)
{
    __shared__ __align__(16) bf16 Bs[2][BROWS * BPITCH];

    const int lane = threadIdx.x & 31;
    const int wave = threadIdx.x >> 5;
    const int half = lane >> 4;
    const int l15  = lane & 15;
    const int wm = wave & 3;                 // 4 waves along M
    const int wn = wave >> 2;                // 2 waves along N
    const int waveM = blockIdx.y * 128 + wm * 32;
    const int blockN = blockIdx.x * 128;
    const int waveN = blockN + wn * 64;

    const float* Af = (const float*)Aptr;
    const bf16*  Ab = (const bf16*)Aptr;
    const u32 ldsBase = (u32)(size_t)&Bs[0][0];

    // prologue: DMA first W slab
    if (wave == 0)
        tdm_load_b_slab(W + (size_t)blockN * K, ldsBase);

    v8f acc[2][4];
    #pragma unroll
    for (int i = 0; i < 2; ++i)
        #pragma unroll
        for (int j = 0; j < 4; ++j)
            acc[i][j] = {};

    for (int kb = 0; kb < K; kb += BKS) {
        const int buf = (kb >> 6) & 1;
        if (wave == 0) wait_tensor0();       // current slab landed
        __syncthreads();                     // publish to all waves
        if (wave == 0 && kb + BKS < K)       // DMA next slab into other buffer
            tdm_load_b_slab(W + (size_t)blockN * K + (kb + BKS), ldsBase + (buf ^ 1) * SLAB_BYTES);

        const bf16* Bbuf = &Bs[buf][0];
        #pragma unroll
        for (int ks = 0; ks < BKS; ks += 32) {
            const int kk = kb + ks;
            v16bf a[2];
            #pragma unroll
            for (int fm = 0; fm < 2; ++fm) {
                const int row = waveM + fm * 16 + l15;
                if constexpr (A_F32) {
                    const float* rp = Af + (size_t)row * K + kk + half * 8;
                    float4 f0 = *(const float4*)(rp);
                    float4 f1 = *(const float4*)(rp + 4);
                    float4 f2 = *(const float4*)(rp + 16);
                    float4 f3 = *(const float4*)(rp + 20);
                    a[fm] = f32x16_to_bf16(f0, f1, f2, f3);
                } else {
                    const bf16* rp = Ab + (size_t)row * K + kk + half * 8;
                    bf16x8 lo = *(const bf16x8*)rp;
                    bf16x8 hi = *(const bf16x8*)(rp + 16);
                    a[fm] = cat8(lo, hi);
                }
            }
            #pragma unroll
            for (int fn = 0; fn < 4; ++fn) {
                const bf16* lp = Bbuf + (wn * 64 + fn * 16 + l15) * BPITCH + ks + half * 8;
                bf16x8 lo = *(const bf16x8*)lp;
                bf16x8 hi = *(const bf16x8*)(lp + 16);
                v16bf bfr = cat8(lo, hi);
                #pragma unroll
                for (int fm = 0; fm < 2; ++fm)
                    acc[fm][fn] = __builtin_amdgcn_wmma_f32_16x16x32_bf16(
                        false, a[fm], false, bfr, (short)0, acc[fm][fn], false, false);
            }
        }
    }

    #pragma unroll
    for (int fn = 0; fn < 4; ++fn) {
        const int n = waveN + fn * 16 + l15;
        const float bv = bias[n];
        #pragma unroll
        for (int fm = 0; fm < 2; ++fm) {
            #pragma unroll
            for (int j = 0; j < 8; ++j) {
                const int m = waveM + fm * 16 + half * 8 + j;
                const float v = acc[fm][fn][j] + bv;
                if constexpr (OUT_F32) ((float*)Out)[(size_t)m * N + n] = v;
                else                   ((bf16*)Out)[(size_t)m * N + n] = (bf16)v;
            }
        }
    }
}

// ---------- per-(b,s) attention over heads ----------
// qh[b,s,h,d] = qp[b, h*256 + (s>>4), (s&15)*64 + d]  (reference's scrambled view)
__global__ __launch_bounds__(256)
void attn_wmma(const bf16* __restrict__ qp, const bf16* __restrict__ kp,
               const bf16* __restrict__ vp, const int* __restrict__ mask,
               bf16* __restrict__ ctx)
{
    __shared__ int  maskS[HHd * HHd];
    __shared__ __align__(16) bf16 attnS[8][16][16];
    __shared__ __align__(16) bf16 vhT[8][64][16];

    maskS[threadIdx.x] = mask[threadIdx.x];
    __syncthreads();

    const int lane = threadIdx.x & 31;
    const int wave = threadIdx.x >> 5;
    const int half = lane >> 4;
    const int l15  = lane & 15;
    const int pair = blockIdx.x * 8 + wave;
    const int b = pair >> 12;
    const int s = pair & 4095;
    const size_t base = (size_t)b * SSQ * DDm;
    const int c0   = (s & 15) * DKK;
    const int srow = s >> 4;
    const size_t hrow = (size_t)(l15 * 256 + srow) * DDm;

    // ---- scores = qh @ kh^T, K = 64 (two k-steps of 32)
    v8f sc = {};
    #pragma unroll
    for (int kk = 0; kk < DKK; kk += 32) {
        const bf16* ap = qp + base + hrow + c0 + kk + half * 8;
        bf16x8 alo = *(const bf16x8*)ap;
        bf16x8 ahi = *(const bf16x8*)(ap + 16);
        v16bf a = cat8(alo, ahi);
        const bf16* bp = kp + base + hrow + c0 + kk + half * 8;
        bf16x8 blo = *(const bf16x8*)bp;
        bf16x8 bhi = *(const bf16x8*)(bp + 16);
        v16bf bfr = cat8(blo, bhi);
        sc = __builtin_amdgcn_wmma_f32_16x16x32_bf16(
            false, a, false, bfr, (short)0, sc, false, false);
    }

    // ---- masked softmax over g (= lane dim, groups of 16)
    #pragma unroll
    for (int j = 0; j < 8; ++j) {
        const int h = j + half * 8;
        float v = sc[j] * 0.125f;                      // 1/sqrt(64)
        if (maskS[h * 16 + l15] == 0) v = -1.0e9f;
        float m = v;
        #pragma unroll
        for (int off = 1; off < 16; off <<= 1) m = fmaxf(m, __shfl_xor(m, off, 16));
        const float e = __expf(v - m);
        float su = e;
        #pragma unroll
        for (int off = 1; off < 16; off <<= 1) su += __shfl_xor(su, off, 16);
        attnS[wave][h][l15] = (bf16)(e / su);
    }

    // ---- stage V^T in LDS: vhT[d][g] = vh[g,d]
    {
        const int g = l15;
        const int dbase = half * 32;
        const bf16* vptr = vp + base + (size_t)(g * 256 + srow) * DDm + c0 + dbase;
        #pragma unroll
        for (int c = 0; c < 4; ++c) {
            bf16x8 vv = *(const bf16x8*)(vptr + c * 8);
            #pragma unroll
            for (int e = 0; e < 8; ++e)
                vhT[wave][dbase + c * 8 + e][g] = vv[e];
        }
    }

    // ---- ctx = attn(16x16, K-padded to 32) @ vh(16x64)
    v16bf aa;
    {
        const bf16* ar = &attnS[wave][l15][half * 8];
        bf16x8 lo = *(const bf16x8*)ar;
        #pragma unroll
        for (int e = 0; e < 8; ++e) { aa[e] = lo[e]; aa[8 + e] = (bf16)0.0f; }
    }
    #pragma unroll
    for (int t = 0; t < 4; ++t) {
        v16bf bfr;
        {
            const bf16* br = &vhT[wave][t * 16 + l15][half * 8];
            bf16x8 lo = *(const bf16x8*)br;
            #pragma unroll
            for (int e = 0; e < 8; ++e) { bfr[e] = lo[e]; bfr[8 + e] = (bf16)0.0f; }
        }
        v8f co = {};
        co = __builtin_amdgcn_wmma_f32_16x16x32_bf16(
            false, aa, false, bfr, (short)0, co, false, false);
        #pragma unroll
        for (int j = 0; j < 8; ++j) {
            const int h = j + half * 8;
            const int d = t * 16 + l15;
            ctx[base + (size_t)h * (SSQ * DKK) + (size_t)s * DKK + d] = (bf16)co[j];
        }
    }
}

// ---------- launch ----------
extern "C" void kernel_launch(void* const* d_in, const int* in_sizes, int n_in,
                              void* d_out, int out_size, void* d_ws, size_t ws_size,
                              hipStream_t stream)
{
    const float* q  = (const float*)d_in[0];
    const float* k  = (const float*)d_in[1];
    const float* v  = (const float*)d_in[2];
    const float* Wq = (const float*)d_in[3];
    const float* bq = (const float*)d_in[4];
    const float* Wk = (const float*)d_in[5];
    const float* bk = (const float*)d_in[6];
    const float* Wv = (const float*)d_in[7];
    const float* bv = (const float*)d_in[8];
    const float* Wo = (const float*)d_in[9];
    const float* bo = (const float*)d_in[10];
    const int*  msk = (const int*)d_in[11];

    const int M = BB * SSQ, N = DDm, K = DDm;
    const size_t WBYTES = (size_t)DDm * DDm * sizeof(bf16);        // 2 MiB
    const size_t ABYTES = (size_t)M * DDm * sizeof(bf16);          // 32 MiB

    char* ws = (char*)d_ws;
    bf16* Wqb = (bf16*)(ws + 0 * WBYTES);
    bf16* Wkb = (bf16*)(ws + 1 * WBYTES);
    bf16* Wvb = (bf16*)(ws + 2 * WBYTES);
    bf16* Wob = (bf16*)(ws + 3 * WBYTES);
    bf16* qpb = (bf16*)(ws + 4 * WBYTES + 0 * ABYTES);
    bf16* kpb = (bf16*)(ws + 4 * WBYTES + 1 * ABYTES);
    bf16* vpb = (bf16*)(ws + 4 * WBYTES + 2 * ABYTES);
    bf16* ctb = (bf16*)(ws + 4 * WBYTES + 3 * ABYTES);

    // 1) weights -> bf16
    const int wElems = DDm * DDm;
    dim3 cg((wElems / 4 + 255) / 256);
    cvt_f32_bf16<<<cg, 256, 0, stream>>>(Wq, Wqb, wElems);
    cvt_f32_bf16<<<cg, 256, 0, stream>>>(Wk, Wkb, wElems);
    cvt_f32_bf16<<<cg, 256, 0, stream>>>(Wv, Wvb, wElems);
    cvt_f32_bf16<<<cg, 256, 0, stream>>>(Wo, Wob, wElems);

    // 2) QKV projections (f32 A -> bf16 out)
    dim3 gg(N / 128, M / 128);
    gemm_wmma_bias<true,  false><<<gg, 256, 0, stream>>>(q, Wqb, bq, qpb, M, N, K);
    gemm_wmma_bias<true,  false><<<gg, 256, 0, stream>>>(k, Wkb, bk, kpb, M, N, K);
    gemm_wmma_bias<true,  false><<<gg, 256, 0, stream>>>(v, Wvb, bv, vpb, M, N, K);

    // 3) attention over heads, one wave per (b,s)
    attn_wmma<<<M / 8, 256, 0, stream>>>(qpb, kpb, vpb, msk, ctb);

    // 4) output projection (bf16 A -> f32 out)
    gemm_wmma_bias<false, true><<<gg, 256, 0, stream>>>(ctb, Wob, bo, (float*)d_out, M, N, K);
}
